// MulRelRanker_26774826123924
// MI455X (gfx1250) — matrix operand order repaired and checked by reading, not compiled, not run on hardware
//
#include <hip/hip_runtime.h>

typedef _Float16 f16;
typedef __attribute__((ext_vector_type(16))) _Float16 v16h;
typedef __attribute__((ext_vector_type(8)))  float    v8f;

#define Mm    400
#define Ww    128
#define Cc    8
#define Ll    50
#define Dd    300
#define Rr    3
#define Hh    100
#define M1    401
#define MC    3208
#define MCP   3216      // 201*16
#define DP    320       // 10*32
#define KBn   10
#define TEE   201       // tiles along MC
#define TEE_NB 67       // N-strips of 3 tiles (201 = 3*67, exact)
#define N1P   416       // 26*16
#define TRC   26        // tiles along M1
#define K3    900
#define NEGC  1.0e10f
#define RSQD  5.7735026919e-2f   // 1/sqrt(300)
#define LAMc  0.3f

// ---------------- WMMA fragment helpers ----------------
__device__ __forceinline__ v16h load_b_frag(const f16* __restrict__ B, int Kp,
                                            int colBase, int k0, int lane) {
  int col = colBase + (lane & 15);
  int grp = (lane >> 4) & 1;
  const f16* p = B + (size_t)col * Kp + (size_t)(k0 + grp * 16);
  return *(const v16h*)p;
}

#define WMMA_F16(A, B, C) \
  __builtin_amdgcn_wmma_f32_16x16x32_f16(false, (A), false, (B), (short)0, (C), false, false)

// ---------------- utility ----------------
__global__ void k_zero(float* p, int n) {
  int i = blockIdx.x * blockDim.x + threadIdx.x;
  if (i < n) p[i] = 0.0f;
}

// ---------------- build padded entity matrix E (MCP x DP), f32 + f16 ----------------
__global__ void k_gatherE(const int* __restrict__ eids, const float* __restrict__ eemb,
                          const float* __restrict__ pad, float* __restrict__ Ef,
                          f16* __restrict__ E16) {
  int row = blockIdx.x;            // 0..MCP-1
  int d   = threadIdx.x;           // 0..DP-1
  float v = 0.0f;
  if (row < MC && d < Dd) {
    int m = row >> 3, c = row & 7;
    if (m < Mm) {
      int id = eids[m * Cc + c];
      v = eemb[(size_t)id * Dd + d];
    } else {
      v = pad[d];
    }
  }
  Ef[(size_t)row * DP + d]  = v;
  E16[(size_t)row * DP + d] = (f16)v;
}

// ---------------- per-mention attention: ctx (M x D) and local (MC) ----------------
__global__ __launch_bounds__(128) void k_ctx(const int* __restrict__ tok_ids,
                     const float* __restrict__ tok_mask,
                     const float* __restrict__ Ef,
                     const float* __restrict__ word_emb,
                     const float* __restrict__ adiag,
                     const float* __restrict__ tdiag,
                     const float* __restrict__ emask,
                     float* __restrict__ ctx_g,
                     float* __restrict__ localb) {
  __shared__ float e_s[Cc * Dd];
  __shared__ float dA[Dd], dT[Dd], ctx_s[Dd];
  __shared__ float ta[Ww];
  __shared__ float rv[Ww]; __shared__ int ri[Ww];
  __shared__ int   top_w[25]; __shared__ int top_id[25];
  __shared__ float top_v[25]; __shared__ float att_p[25];

  int m = blockIdx.x, t = threadIdx.x;
  for (int i = t; i < Cc * Dd; i += 128)
    e_s[i] = Ef[((size_t)m * Cc + (i / Dd)) * DP + (i % Dd)];
  for (int i = t; i < Dd; i += 128) { dA[i] = adiag[i]; dT[i] = tdiag[i]; }
  __syncthreads();

  // attention scores (one w per thread), tok_att = max over c
  {
    int w = t;  // blockDim == Ww == 128
    int id = tok_ids[m * Ww + w];
    const float* wp = word_emb + (size_t)id * Dd;
    float acc0=0,acc1=0,acc2=0,acc3=0,acc4=0,acc5=0,acc6=0,acc7=0;
    for (int d = 0; d < Dd; ++d) {
      float ad = wp[d] * dA[d];
      acc0 += e_s[0*Dd+d]*ad; acc1 += e_s[1*Dd+d]*ad;
      acc2 += e_s[2*Dd+d]*ad; acc3 += e_s[3*Dd+d]*ad;
      acc4 += e_s[4*Dd+d]*ad; acc5 += e_s[5*Dd+d]*ad;
      acc6 += e_s[6*Dd+d]*ad; acc7 += e_s[7*Dd+d]*ad;
    }
    float tm = tok_mask[m * Ww + w];
    float bias = (tm - 1.0f) * NEGC;
    float best = -3.4e38f;
    best = fmaxf(best, acc0*tm+bias); best = fmaxf(best, acc1*tm+bias);
    best = fmaxf(best, acc2*tm+bias); best = fmaxf(best, acc3*tm+bias);
    best = fmaxf(best, acc4*tm+bias); best = fmaxf(best, acc5*tm+bias);
    best = fmaxf(best, acc6*tm+bias); best = fmaxf(best, acc7*tm+bias);
    ta[w] = best;
  }
  __syncthreads();

  // top-25 selection (iterative argmax)
  for (int j = 0; j < 25; ++j) {
    rv[t] = ta[t]; ri[t] = t; __syncthreads();
    for (int s = 64; s > 0; s >>= 1) {
      if (t < s) { if (rv[t+s] > rv[t]) { rv[t] = rv[t+s]; ri[t] = ri[t+s]; } }
      __syncthreads();
    }
    if (t == 0) { top_v[j] = rv[0]; top_w[j] = ri[0]; ta[ri[0]] = -3.4e38f; }
    __syncthreads();
  }
  if (t == 0) {
    float mx = top_v[0], s = 0.0f;
    for (int j = 0; j < 25; ++j) s += expf(top_v[j] - mx);
    float inv = 1.0f / s;
    for (int j = 0; j < 25; ++j) {
      att_p[j]  = expf(top_v[j] - mx) * inv;
      top_id[j] = tok_ids[m * Ww + top_w[j]];
    }
  }
  __syncthreads();

  // ctx vector
  for (int d = t; d < Dd; d += 128) {
    float s = 0.0f;
    for (int j = 0; j < 25; ++j)
      s += word_emb[(size_t)top_id[j] * Dd + d] * att_p[j];
    float v = s * dT[d];
    ctx_s[d] = v;
    ctx_g[(size_t)m * Dd + d] = v;
  }
  __syncthreads();

  // local[m,c] = <ent_vec, ctx> with mask
  {
    int c = t >> 4, l = t & 15;
    float p = 0.0f;
    for (int d = l; d < Dd; d += 16) p += e_s[c * Dd + d] * ctx_s[d];
    for (int off = 8; off > 0; off >>= 1) p += __shfl_down(p, off, 16);
    if (l == 0) {
      float em = emask[m * Cc + c];
      localb[m * Cc + c] = p * em + (em - 1.0f) * NEGC;
    }
  }
}

// ---------------- bag-of-words (M x 900) ----------------
__global__ void k_bow(const int* lid, const float* lmk,
                      const int* mid, const float* mmk,
                      const int* rid, const float* rmk,
                      const float* __restrict__ snd, float* __restrict__ bow) {
  int m = blockIdx.x, d = threadIdx.x;
  if (d >= Dd) return;
  const int*   ids[3]  = { lid, mid, rid };
  const float* mks[3]  = { lmk, mmk, rmk };
  for (int seg = 0; seg < 3; ++seg) {
    float num = 0.0f, den = 0.0f;
    for (int l = 0; l < Ll; ++l) {
      float mk = mks[seg][m * Ll + l];
      int   id = ids[seg][m * Ll + l];
      num += snd[(size_t)id * Dd + d] * mk;
      den += mk;
    }
    bow[(size_t)m * K3 + seg * Dd + d] = num / (den + 1e-5f);
  }
}

// ---------------- ctx_vecs = tanh(bow @ ctx_W + b)  (400x300, K=900) ----------------
__global__ void k_cvgemm(const float* __restrict__ bow, const float* __restrict__ Wm,
                         const float* __restrict__ bvec, float* __restrict__ cv) {
  int m = blockIdx.x, d = threadIdx.x;
  if (d >= Dd) return;
  float acc = 0.0f;
  const float* br = bow + (size_t)m * K3;
  for (int k = 0; k < K3; ++k) acc += br[k] * Wm[(size_t)k * Dd + d];
  cv[(size_t)m * Dd + d] = tanhf(acc + bvec[d]);
}

// ---------------- pad ctx_vecs to (N1P x DP), f32 + f16 ----------------
__global__ void k_cvpad(const float* __restrict__ cv, const float* __restrict__ padv,
                        float* __restrict__ cvp, f16* __restrict__ cv16) {
  int r = blockIdx.x, d = threadIdx.x;
  float v = 0.0f;
  if (d < Dd) {
    if (r < Mm)       v = cv[(size_t)r * Dd + d];
    else if (r == Mm) v = padv[d];
  }
  cvp[(size_t)r * DP + d]  = v;
  cv16[(size_t)r * DP + d] = (f16)v;
}

// ---------------- pack diag-scaled A fragments into WMMA 16-bit A layout ----------------
// grid: (tiles*KBn, R) x 32; dst per-r pitch = tilesKB*512 halves
__global__ __launch_bounds__(32) void k_packA(const float* __restrict__ src, int stride,
                                              const float* __restrict__ scaleBase,
                                              f16* __restrict__ dst, int tilesKB) {
  int bx = blockIdx.x, r = blockIdx.y, lane = threadIdx.x;
  int tile = bx / KBn, kb = bx % KBn;
  const float* sc = scaleBase + r * Dd;
  int grp = lane >> 4, row = tile * 16 + (lane & 15);
  v16h o;
  #pragma unroll
  for (int i = 0; i < 16; ++i) {
    int v = i >> 1, p = i & 1;
    int kl = 16 * grp + 2 * v + p;
    int k = kb * 32 + kl;
    float s = (k < Dd) ? sc[k] : 0.0f;
    o[i] = (f16)(src[(size_t)row * stride + k] * s);
  }
  f16* dp = dst + (size_t)r * tilesKB * 512 + ((size_t)bx * 32 + lane) * 16;
  *(v16h*)dp = o;
}

// ---------------- rcc logits via WMMA; all 3 r share one B load ----------------
__global__ __launch_bounds__(32) void k_rcc(const f16* __restrict__ Apack,
                                            const f16* __restrict__ B16,
                                            float* __restrict__ probs) {
  int tm = blockIdx.x, tn = blockIdx.y, lane = threadIdx.x;
  v8f c0 = {0.f,0.f,0.f,0.f,0.f,0.f,0.f,0.f};
  v8f c1 = c0, c2 = c0;
  const size_t rp = (size_t)TRC * KBn * 512;
  const f16* Ap0 = Apack + (size_t)tm * KBn * 512;
  const f16* Ap1 = Ap0 + rp;
  const f16* Ap2 = Ap1 + rp;
  for (int kb = 0; kb < KBn; ++kb) {
    v16h b  = load_b_frag(B16, DP, tn * 16, kb * 32, lane);
    v16h a0 = *(const v16h*)(Ap0 + (size_t)kb * 512 + (size_t)lane * 16);
    v16h a1 = *(const v16h*)(Ap1 + (size_t)kb * 512 + (size_t)lane * 16);
    v16h a2 = *(const v16h*)(Ap2 + (size_t)kb * 512 + (size_t)lane * 16);
    c0 = WMMA_F16(a0, b, c0);
    c1 = WMMA_F16(a1, b, c1);
    c2 = WMMA_F16(a2, b, c2);
  }
  int col   = tn * 16 + (lane & 15);
  int rbase = tm * 16 + ((lane < 16) ? 0 : 8);
  #pragma unroll
  for (int j = 0; j < 8; ++j) {
    int row = rbase + j;
    if (row < M1 && col < M1) {
      float dmask = (row == col) ? (-2.0f * NEGC) : 0.0f;
      size_t o = (size_t)row * M1 + col;
      probs[0 * (size_t)(M1 * M1) + o] = (c0[j] + dmask) * RSQD;
      probs[1 * (size_t)(M1 * M1) + o] = (c1[j] + dmask) * RSQD;
      probs[2 * (size_t)(M1 * M1) + o] = (c2[j] + dmask) * RSQD;
    }
  }
}

// ---------------- softmax over n for each (r,m) row of length 401 ----------------
__global__ void k_softmax401(float* __restrict__ probs) {
  __shared__ float red[256];
  int t = threadIdx.x;
  float* base = probs + (size_t)blockIdx.x * M1;
  float mx = -3.4e38f;
  for (int i = t; i < M1; i += 256) mx = fmaxf(mx, base[i]);
  red[t] = mx; __syncthreads();
  for (int s = 128; s > 0; s >>= 1) { if (t < s) red[t] = fmaxf(red[t], red[t+s]); __syncthreads(); }
  mx = red[0]; __syncthreads();
  float sm = 0.0f;
  for (int i = t; i < M1; i += 256) sm += expf(base[i] - mx);
  red[t] = sm; __syncthreads();
  for (int s = 128; s > 0; s >>= 1) { if (t < s) red[t] += red[t+s]; __syncthreads(); }
  float inv = 1.0f / red[0];
  for (int i = t; i < M1; i += 256) base[i] = expf(base[i] - mx) * inv;
}

// ---------------- weights = probs + probs^T ----------------
__global__ void k_weights(const float* __restrict__ probs, float* __restrict__ wts) {
  int idx = blockIdx.x * blockDim.x + threadIdx.x;
  if (idx >= Rr * M1 * M1) return;
  int r = idx / (M1 * M1), rem = idx % (M1 * M1);
  int m = rem / M1, n = rem % M1;
  wts[idx] = probs[idx] + probs[(size_t)r * (M1 * M1) + (size_t)n * M1 + m];
}

// ---------------- ee via WMMA: 16x48 strip per wave, 3 r x 3 n-tiles = 9 WMMA / K-step ----------------
__global__ __launch_bounds__(32) void k_ee(const f16* __restrict__ Apack,
                                           const f16* __restrict__ B16,
                                           const float* __restrict__ wts,
                                           const float* __restrict__ emask,
                                           float* __restrict__ ee) {
  int tm  = blockIdx.x;           // 0..200
  int tn0 = blockIdx.y * 3;       // 0..198, exact (201 = 3*67)
  int lane = threadIdx.x;
  v8f z = {0.f,0.f,0.f,0.f,0.f,0.f,0.f,0.f};
  v8f c00 = z, c01 = z, c02 = z;  // r0 x {n0,n1,n2}
  v8f c10 = z, c11 = z, c12 = z;  // r1
  v8f c20 = z, c21 = z, c22 = z;  // r2
  const size_t rp = (size_t)TEE * KBn * 512;
  const f16* Ap0 = Apack + (size_t)tm * KBn * 512;
  const f16* Ap1 = Ap0 + rp;
  const f16* Ap2 = Ap1 + rp;
  for (int kb = 0; kb < KBn; ++kb) {
    v16h b0 = load_b_frag(B16, DP, (tn0 + 0) * 16, kb * 32, lane);
    v16h b1 = load_b_frag(B16, DP, (tn0 + 1) * 16, kb * 32, lane);
    v16h b2 = load_b_frag(B16, DP, (tn0 + 2) * 16, kb * 32, lane);
    v16h a0 = *(const v16h*)(Ap0 + (size_t)kb * 512 + (size_t)lane * 16);
    v16h a1 = *(const v16h*)(Ap1 + (size_t)kb * 512 + (size_t)lane * 16);
    v16h a2 = *(const v16h*)(Ap2 + (size_t)kb * 512 + (size_t)lane * 16);
    if (kb + 1 < KBn)
      __builtin_prefetch(Ap0 + (size_t)(kb + 1) * 512 + (size_t)lane * 16, 0, 1);
    c00 = WMMA_F16(a0, b0, c00); c01 = WMMA_F16(a0, b1, c01); c02 = WMMA_F16(a0, b2, c02);
    c10 = WMMA_F16(a1, b0, c10); c11 = WMMA_F16(a1, b1, c11); c12 = WMMA_F16(a1, b2, c12);
    c20 = WMMA_F16(a2, b0, c20); c21 = WMMA_F16(a2, b1, c21); c22 = WMMA_F16(a2, b2, c22);
  }
  int rbase = tm * 16 + ((lane < 16) ? 0 : 8);
  #pragma unroll
  for (int nt = 0; nt < 3; ++nt) {
    int col = (tn0 + nt) * 16 + (lane & 15);
    int n = col >> 3; if (n > Mm) n = Mm;
    int k = col & 7;
    float em = (n < Mm) ? emask[n * Cc + k] : ((k == 0) ? 1.0f : 0.0f);
    float mb = (em - 1.0f) * NEGC;
    v8f r0 = (nt == 0) ? c00 : (nt == 1) ? c01 : c02;
    v8f r1 = (nt == 0) ? c10 : (nt == 1) ? c11 : c12;
    v8f r2 = (nt == 0) ? c20 : (nt == 1) ? c21 : c22;
    #pragma unroll
    for (int j = 0; j < 8; ++j) {
      int row = rbase + j;
      int m = row >> 3; if (m > Mm) m = Mm;
      size_t wb = (size_t)m * M1 + n;
      float s = wts[0 * (M1 * M1) + wb] * (r0[j] * em + mb)
              + wts[1 * (M1 * M1) + wb] * (r1[j] * em + mb)
              + wts[2 * (M1 * M1) + wb] * (r2[j] * em + mb);
      ee[(size_t)row * MCP + col] = s * (1.0f / (float)Rr);
    }
  }
}

// ---------------- LBP loop ----------------
__global__ void k_rowsum(const float* __restrict__ prev, float* __restrict__ rs) {
  int id = blockIdx.x * blockDim.x + threadIdx.x;
  if (id >= MC) return;
  const float* p = prev + (size_t)id * M1;
  float s = 0.0f;
  for (int j = 0; j < M1; ++j) s += p[j];
  rs[id] = s;
}

__global__ void k_step(const float* __restrict__ ee, const float* __restrict__ localb,
                       const float* __restrict__ rs, const float* __restrict__ prev,
                       float* __restrict__ nxt) {
  int m = blockIdx.x;
  for (int n = threadIdx.x; n < M1; n += blockDim.x) {
    float vm[Cc];
    #pragma unroll
    for (int c = 0; c < Cc; ++c) {
      const float* er = ee + (size_t)(m * Cc + c) * MCP + n * Cc;
      float best = -3.4e38f;
      #pragma unroll
      for (int k = 0; k < Cc; ++k) {
        int nk = n * Cc + k;
        float v = er[k] + localb[nk] + rs[nk] - prev[(size_t)nk * M1 + m];
        best = fmaxf(best, v);
      }
      vm[c] = best;
    }
    float mx = -3.4e38f;
    #pragma unroll
    for (int c = 0; c < Cc; ++c) mx = fmaxf(mx, vm[c]);
    float sm = 0.0f;
    #pragma unroll
    for (int c = 0; c < Cc; ++c) { vm[c] = expf(vm[c] - mx); sm += vm[c]; }
    float inv = 1.0f / sm;
    #pragma unroll
    for (int c = 0; c < Cc; ++c) {
      size_t idx = (size_t)(m * Cc + c) * M1 + n;
      float pv = prev[idx];
      nxt[idx] = logf(vm[c] * inv * LAMc + expf(pv) * (1.0f - LAMc));
    }
  }
}

// ---------------- ent_scores = softmax_c(local + sum_{n!=m} msgs) ----------------
__global__ void k_entscores(const float* __restrict__ localb, const float* __restrict__ msgs,
                            float* __restrict__ out_es) {
  int gid = blockIdx.x * blockDim.x + threadIdx.x;
  if (gid >= MC) return;
  int m = gid >> 3;
  float s = localb[gid];
  const float* mr = msgs + (size_t)gid * M1;
  for (int n = 0; n < M1; ++n) if (n != m) s += mr[n];
  float mx = s;
  for (int off = 4; off > 0; off >>= 1) mx = fmaxf(mx, __shfl_xor(mx, off, 8));
  float e = expf(s - mx), sm = e;
  for (int off = 4; off > 0; off >>= 1) sm += __shfl_xor(sm, off, 8);
  out_es[gid] = e / sm;
}

// ---------------- final 2->100->1 MLP ----------------
__global__ void k_mlp(const float* __restrict__ pem,
                      const float* __restrict__ W1, const float* __restrict__ b1,
                      const float* __restrict__ W2, const float* __restrict__ b2,
                      const float* __restrict__ es, float* __restrict__ out_scores) {
  int gid = blockIdx.x * blockDim.x + threadIdx.x;
  if (gid >= MC) return;
  int m = gid >> 3, c = gid & 7;
  float x0 = es[gid];
  float pe = (m < Mm) ? pem[gid] : ((c == 0) ? 1.0f : 0.0f);
  float x1 = logf(pe + 1e-20f);
  float acc = b2[0];
  for (int h = 0; h < Hh; ++h) {
    float t = x0 * W1[h] + x1 * W1[Hh + h] + b1[h];
    acc += fmaxf(t, 0.0f) * W2[h];
  }
  if (m < Mm) out_scores[gid] = acc;  // scores[:-1]
}

// =====================================================================
extern "C" void kernel_launch(void* const* d_in, const int* in_sizes, int n_in,
                              void* d_out, int out_size, void* d_ws, size_t ws_size,
                              hipStream_t stream) {
  (void)in_sizes; (void)n_in; (void)out_size; (void)ws_size;
  const int*   token_ids  = (const int*)  d_in[0];
  const float* tok_mask   = (const float*)d_in[1];
  const int*   entity_ids = (const int*)  d_in[2];
  const float* emask      = (const float*)d_in[3];
  const float* p_e_m      = (const float*)d_in[4];
  const int*   lid        = (const int*)  d_in[5];
  const float* lmk        = (const float*)d_in[6];
  const int*   rid        = (const int*)  d_in[7];
  const float* rmk        = (const float*)d_in[8];
  const int*   mid        = (const int*)  d_in[9];
  const float* mmk        = (const float*)d_in[10];
  const float* word_emb   = (const float*)d_in[11];
  const float* entity_emb = (const float*)d_in[12];
  const float* snd_emb    = (const float*)d_in[13];
  const float* adiag      = (const float*)d_in[14];
  const float* tdiag      = (const float*)d_in[15];
  const float* rel_embs   = (const float*)d_in[16];
  const float* ew_embs    = (const float*)d_in[17];
  const float* pad_ent    = (const float*)d_in[18];
  const float* pad_ctx    = (const float*)d_in[19];
  const float* ctx_W      = (const float*)d_in[20];
  const float* ctx_b      = (const float*)d_in[21];
  const float* sc_W1      = (const float*)d_in[22];
  const float* sc_b1      = (const float*)d_in[23];
  const float* sc_W2      = (const float*)d_in[24];
  const float* sc_b2      = (const float*)d_in[25];
  float* out = (float*)d_out;

  // ---- workspace carve-up (256B aligned) ----
  char* base = (char*)d_ws; size_t off = 0;
  auto alloc = [&](size_t bytes) -> void* {
    void* p = base + off; off += (bytes + 255) & ~(size_t)255; return p;
  };
  float* Ef     = (float*)alloc((size_t)MCP * DP * 4);
  f16*   E16    = (f16*)  alloc((size_t)MCP * DP * 2);
  float* ctxb   = (float*)alloc((size_t)Mm * Dd * 4);
  float* localb = (float*)alloc((size_t)MCP * 4);
  float* cv     = (float*)alloc((size_t)Mm * Dd * 4);
  float* cvp    = (float*)alloc((size_t)N1P * DP * 4);
  f16*   cv16   = (f16*)  alloc((size_t)N1P * DP * 2);
  float* bow    = (float*)alloc((size_t)Mm * K3 * 4);
  float* probs  = (float*)alloc((size_t)Rr * M1 * M1 * 4);
  float* wts    = (float*)alloc((size_t)Rr * M1 * M1 * 4);
  float* ee     = (float*)alloc((size_t)MCP * MCP * 4);
  float* prevA  = (float*)alloc((size_t)MC * M1 * 4);
  float* prevB  = (float*)alloc((size_t)MC * M1 * 4);
  float* rsum   = (float*)alloc((size_t)MCP * 4);
  f16*   ApEE   = (f16*)  alloc((size_t)Rr * TEE * KBn * 512 * 2);
  f16*   ApRC   = (f16*)  alloc((size_t)Rr * TRC * KBn * 512 * 2);

  // ---- init ----
  k_zero<<<(MCP + 255) / 256, 256, 0, stream>>>(localb, MCP);
  k_zero<<<(MC * M1 + 255) / 256, 256, 0, stream>>>(prevA, MC * M1);

  // ---- front end ----
  k_gatherE<<<MCP, DP, 0, stream>>>(entity_ids, entity_emb, pad_ent, Ef, E16);
  k_ctx<<<Mm, 128, 0, stream>>>(token_ids, tok_mask, Ef, word_emb, adiag, tdiag,
                                emask, ctxb, localb);
  k_bow<<<Mm, DP, 0, stream>>>(lid, lmk, mid, mmk, rid, rmk, snd_emb, bow);
  k_cvgemm<<<Mm, DP, 0, stream>>>(bow, ctx_W, ctx_b, cv);
  k_cvpad<<<N1P, DP, 0, stream>>>(cv, pad_ctx, cvp, cv16);

  // ---- pack WMMA A fragments (diag-scaled by ew / rel per r) ----
  k_packA<<<dim3(TRC * KBn, Rr), 32, 0, stream>>>(cvp, DP, ew_embs, ApRC, TRC * KBn);
  k_packA<<<dim3(TEE * KBn, Rr), 32, 0, stream>>>(Ef,  DP, rel_embs, ApEE, TEE * KBn);

  // ---- rcc -> softmax -> weights ----
  k_rcc<<<dim3(TRC, TRC), 32, 0, stream>>>(ApRC, cv16, probs);
  k_softmax401<<<Rr * M1, 256, 0, stream>>>(probs);
  k_weights<<<(Rr * M1 * M1 + 255) / 256, 256, 0, stream>>>(probs, wts);

  // ---- dominant GEMM: ee (16x48 strip per wave, 9 WMMA per K-step) ----
  k_ee<<<dim3(TEE, TEE_NB), 32, 0, stream>>>(ApEE, E16, wts, emask, ee);

  // ---- 10 LBP iterations ----
  float* cur = prevA; float* nxt = prevB;
  for (int it = 0; it < 10; ++it) {
    k_rowsum<<<(MC + 255) / 256, 256, 0, stream>>>(cur, rsum);
    k_step<<<M1, 256, 0, stream>>>(ee, localb, rsum, cur, nxt);
    float* tmp = cur; cur = nxt; nxt = tmp;
  }

  // ---- epilogue: ent_scores into out[3200..], scores into out[0..3199] ----
  k_entscores<<<(MC + 255) / 256, 256, 0, stream>>>(localb, cur, out + Mm * Cc);
  k_mlp<<<(MC + 255) / 256, 256, 0, stream>>>(p_e_m, sc_W1, sc_b1, sc_W2, sc_b2,
                                              out + Mm * Cc, out);
}
